// ViLLayer_25280177504427
// MI455X (gfx1250) — matrix operand order repaired and checked by reading, not compiled, not run on hardware
//
#include <hip/hip_runtime.h>

// ---------------- problem constants ----------------
#define BB     4
#define CDIM   256
#define LSEQ   4096
#define TTOK   (BB * LSEQ)      // 16384 tokens
#define DINNER 512
#define NST    16
#define RK     16
#define SCHUNK 64               // scan tokens staged per TDM transfer

typedef __attribute__((ext_vector_type(8)))  float        v8f;
typedef __attribute__((ext_vector_type(16))) __bf16       v16bf;
typedef __attribute__((ext_vector_type(8)))  __bf16       v8bf;
typedef __attribute__((ext_vector_type(4)))  unsigned int u32x4;
typedef __attribute__((ext_vector_type(8)))  unsigned int u32x8;

// ---------------- helpers: WMMA fragment loads (16-bit layouts, ISA 7.12.2) ----
// A 16x32 bf16: lanes 0-15 hold row M=lane, K = k0..k0+15;
//               lanes 16-31 hold row M=lane-16, K = k0+16..k0+31 (contiguous in K).
__device__ __forceinline__ v16bf load_a_frag(const __bf16* __restrict__ A, int lda,
                                             int row0, int k0, int lane) {
  int m  = row0 + (lane & 15);
  int kb = k0 + ((lane >> 4) << 4);           // +0 or +16
  const __bf16* p = A + m * lda + kb;
  v8bf lo = *(const v8bf*)(p);
  v8bf hi = *(const v8bf*)(p + 8);
  v16bf f;
#pragma unroll
  for (int i = 0; i < 8; i++) { f[i] = lo[i]; f[i + 8] = hi[i]; }
  return f;
}

// B 32x16 bf16 from row-major weights W[N,K]: lane holds column n = lane&15,
// lanes 0-15 K = k0..k0+15, lanes 16-31 K = k0+16..k0+31 (contiguous in K).
__device__ __forceinline__ v16bf load_b_frag(const __bf16* __restrict__ W, int ldw,
                                             int n0, int k0, int lane) {
  int n  = n0 + (lane & 15);
  int kb = k0 + ((lane >> 4) << 4);           // +0 or +16
  return *(const v16bf*)(W + n * ldw + kb);
}

// One wave computes NT adjacent 16x16 output tiles, reusing the A fragment
// across NT WMMAs per k-step (cuts redundant A traffic NT x).
template <int NT>
__device__ __forceinline__ void gemm_multi(const __bf16* __restrict__ A, int lda,
                                           const __bf16* __restrict__ W, int ldw,
                                           int row0, int n0, int K, int lane, v8f* acc) {
  for (int k0 = 0; k0 < K; k0 += 32) {
    v16bf a = load_a_frag(A, lda, row0, k0, lane);
#pragma unroll
    for (int t = 0; t < NT; t++) {
      v16bf b = load_b_frag(W, ldw, n0 + 16 * t, k0, lane);
      acc[t] = __builtin_amdgcn_wmma_f32_16x16x32_bf16(false, a, false, b, (short)0,
                                                       acc[t], false, false);
    }
  }
}

// ---------------- TDM: 2D tile (48 f32 x rows) global -> LDS ---------------
// D# per CDNA5 ISA: group0 = {count=1, lds_addr,
// global_addr, type=2}; group1 = {data_size=4B, tensor 48 x rows, tile 48 x rows,
// dim0 stride 48}. Two-group form (VADDR2/3 = NULL, tensors up to 2D).
__device__ __forceinline__ void tdm_load_rows(const float* gsrc, unsigned lds_off,
                                              unsigned rows) {
  unsigned long long ga = (unsigned long long)(size_t)gsrc;
  u32x4 g0;
  g0[0] = 1u;                                            // count=1
  g0[1] = lds_off;                                       // lds_addr
  g0[2] = (unsigned)ga;                                  // global_addr[31:0]
  g0[3] = (unsigned)((ga >> 32) & 0x1FFFFFFu) | (2u << 30); // ga[56:32] | type=2
  u32x8 g1;
  g1[0] = 2u << 16;                // workgroup_mask=0, data_size=2 (4 bytes)
  g1[1] = 48u << 16;               // tensor_dim0[15:0] = 48
  g1[2] = rows << 16;              // tensor_dim0 hi=0, tensor_dim1[15:0] = rows
  g1[3] = 48u << 16;               // tensor_dim1 hi=0, tile_dim0 = 48
  g1[4] = rows & 0xFFFFu;          // tile_dim1 = rows, tile_dim2 = 0
  g1[5] = 48u;                     // tensor_dim0_stride = 48
  g1[6] = 0u;
  g1[7] = 0u;
  asm volatile("tensor_load_to_lds %0, %1" :: "s"(g0), "s"(g1) : "memory");
}

// ---------------- 0: weight f32 -> bf16 ----------------
__global__ void cvt_f32_bf16(const float* __restrict__ src, __bf16* __restrict__ dst, int n) {
  int i = blockIdx.x * 256 + threadIdx.x;
  if (i < n) dst[i] = (__bf16)src[i];
}

__global__ void zero_f32(float* __restrict__ p, int n) {
  int i = blockIdx.x * 256 + threadIdx.x;
  if (i < n) p[i] = 0.f;
}

// ---------------- 1: fused double LayerNorm, [B,C,L] -> bf16 [T,C] -----------
__global__ void ln2_kernel(const float* __restrict__ x, const float* __restrict__ w1,
                           const float* __restrict__ w2, const float* __restrict__ b2,
                           __bf16* __restrict__ xn) {
  __shared__ float lds[8448];                 // [j:32][c:256] with stride 264
  int b = blockIdx.y;
  int l0 = blockIdx.x * 32;
  int tid = threadIdx.x;
#pragma unroll
  for (int i = 0; i < 32; i++) {
    int f = i * 256 + tid;
    int c = f >> 5, j = f & 31;
    lds[j * 264 + c] = x[(b * 256 + c) * 4096 + l0 + j];
  }
  __syncthreads();
  int j = tid >> 3;                           // token within tile
  int p = tid & 7;                            // 8 lanes cooperate per token
  float v[32];
#pragma unroll
  for (int i = 0; i < 32; i++) v[i] = lds[j * 264 + p * 32 + i];

  float s = 0.f;
#pragma unroll
  for (int i = 0; i < 32; i++) s += v[i];
  s += __shfl_xor(s, 1); s += __shfl_xor(s, 2); s += __shfl_xor(s, 4);
  float mu = s * (1.f / 256.f);
  float vr = 0.f;
#pragma unroll
  for (int i = 0; i < 32; i++) { float d = v[i] - mu; vr += d * d; }
  vr += __shfl_xor(vr, 1); vr += __shfl_xor(vr, 2); vr += __shfl_xor(vr, 4);
  float rs = rsqrtf(vr * (1.f / 256.f) + 1e-5f);
#pragma unroll
  for (int i = 0; i < 32; i++) v[i] = (v[i] - mu) * rs * w1[p * 32 + i];

  s = 0.f;
#pragma unroll
  for (int i = 0; i < 32; i++) s += v[i];
  s += __shfl_xor(s, 1); s += __shfl_xor(s, 2); s += __shfl_xor(s, 4);
  float mu2 = s * (1.f / 256.f);
  vr = 0.f;
#pragma unroll
  for (int i = 0; i < 32; i++) { float d = v[i] - mu2; vr += d * d; }
  vr += __shfl_xor(vr, 1); vr += __shfl_xor(vr, 2); vr += __shfl_xor(vr, 4);
  float rs2 = rsqrtf(vr * (1.f / 256.f) + 1e-5f);

  int t = b * 4096 + l0 + j;
#pragma unroll
  for (int i = 0; i < 32; i++) {
    int ch = p * 32 + i;
    xn[t * 256 + ch] = (__bf16)((v[i] - mu2) * rs2 * w2[ch] + b2[ch]);
  }
}

// ---------------- 2: in_proj GEMM  xz[T,1024] = xn[T,256] * W[1024,256]^T ----
// 8 waves/block, each wave owns 4 N-tiles (A reuse).
__global__ void inproj_gemm(const __bf16* __restrict__ xn, const __bf16* __restrict__ W,
                            __bf16* __restrict__ xz) {
  int lane = threadIdx.x & 31, wave = threadIdx.x >> 5;
  int row0 = blockIdx.x * 16;
  int n0 = (blockIdx.y * 8 + wave) * 64;
  v8f acc[4] = {};
  gemm_multi<4>(xn, 256, W, 256, row0, n0, 256, lane, acc);
  int rb = (lane >> 4) * 8;
#pragma unroll
  for (int t = 0; t < 4; t++) {
    int col = n0 + 16 * t + (lane & 15);
#pragma unroll
    for (int r = 0; r < 8; r++)
      xz[(row0 + rb + r) * 1024 + col] = (__bf16)acc[t][r];
  }
}

// ---------------- 3: depthwise causal conv (k=4) + bias + SiLU --------------
__global__ void conv_silu(const __bf16* __restrict__ xz, const float* __restrict__ cw,
                          const float* __restrict__ cb, __bf16* __restrict__ uact) {
  int idx = blockIdx.x * 256 + threadIdx.x;   // idx = t*512 + d
  int d = idx & 511, t = idx >> 9;
  int l = t & 4095, base = t - l;
  float acc = cb[d];
#pragma unroll
  for (int k = 0; k < 4; k++) {
    int lk = l - 3 + k;
    if (lk >= 0) acc += cw[d * 4 + k] * (float)xz[(base + lk) * 1024 + d];
  }
  float sv = acc * (1.f / (1.f + __expf(-acc)));
  uact[idx] = (__bf16)sv;
}

// ---------------- 4: x_proj GEMM  dbl[T,48] = uact[T,512] * W[48,512]^T -----
// One wave per 16-token row, all 3 N-tiles with A reuse.
__global__ void xproj_gemm(const __bf16* __restrict__ uact, const __bf16* __restrict__ W,
                           float* __restrict__ dbl) {
  int lane = threadIdx.x;
  int row0 = blockIdx.x * 16;
  v8f acc[3] = {};
  gemm_multi<3>(uact, 512, W, 512, row0, 0, 512, lane, acc);
  int rb = (lane >> 4) * 8;
#pragma unroll
  for (int t = 0; t < 3; t++) {
    int col = 16 * t + (lane & 15);
#pragma unroll
    for (int r = 0; r < 8; r++)
      dbl[(row0 + rb + r) * 48 + col] = acc[t][r];
  }
}

// ---------------- 5: dt_proj (K=16 zero-padded to 32) + softplus ------------
// One wave owns 4 N-tiles, single k-step, A reuse across the 4 WMMAs.
__global__ void dtproj_gemm(const float* __restrict__ dbl, const __bf16* __restrict__ Wdt,
                            const float* __restrict__ bias, float* __restrict__ delta) {
  int lane = threadIdx.x;
  int row0 = blockIdx.x * 16, n0 = blockIdx.y * 64;
  int half = lane >> 4;
  // A frag: valid K are 0..15; half0 holds K0..15,
  // half1 holds zeros for K16..31.
  const float* ap = dbl + (row0 + (lane & 15)) * 48;
  v16bf a;
#pragma unroll
  for (int i = 0; i < 8; i++) { a[i] = half == 0 ? (__bf16)ap[i] : (__bf16)0.f; a[i + 8] = half == 0 ? (__bf16)ap[i + 8] : (__bf16)0.f; }
  int rb = half * 8;
#pragma unroll
  for (int t = 0; t < 4; t++) {
    // B frag: lanes 0-15 hold K0..15 (valid); lanes 16-31 hold K16..31 -> zero.
    v16bf b = {};
    if (half == 0) b = *(const v16bf*)(Wdt + (n0 + 16 * t + (lane & 15)) * 16);
    v8f acc = {};
    acc = __builtin_amdgcn_wmma_f32_16x16x32_bf16(false, a, false, b, (short)0, acc,
                                                  false, false);
    int col = n0 + 16 * t + (lane & 15);
    float bi = bias[col];
#pragma unroll
    for (int r = 0; r < 8; r++) {
      float v = acc[r] + bi;
      float sp = (v > 20.f) ? v : __logf(1.f + __expf(v));   // softplus
      delta[(row0 + rb + r) * 512 + col] = sp;
    }
  }
}

// ---------------- 6: selective scan (sequential over L) ---------------------
// One lane owns one (b,d) channel; 16 SSM states live in VGPRs. Per-token B/C
// rows are staged 64 tokens at a time by the Tensor Data Mover into a
// double-buffered LDS tile: the DMA for chunk c+1 overlaps compute on chunk c,
// and the on-critical-path B/C reads become ds_loads. delta stream is
// prefetched; y overwrites delta in place.
__global__ void scan_kernel(const float* __restrict__ A_log, const __bf16* __restrict__ uact,
                            const float* __restrict__ dbl, float* __restrict__ dy) {
  __shared__ float sdbl[2][SCHUNK * 48];
  int b = blockIdx.x >> 1;
  int d = (blockIdx.x & 1) * 256 + threadIdx.x;
  bool issuer = (threadIdx.x < 32);           // wave 0 drives the TDM
  float A[NST], h[NST];
#pragma unroll
  for (int n = 0; n < NST; n++) { A[n] = -__expf(A_log[d * NST + n]); h[n] = 0.f; }
  int t0 = b * 4096;
  const int nch = 4096 / SCHUNK;
  if (issuer)                                  // prologue: stage chunk 0
    tdm_load_rows(dbl + (size_t)t0 * 48, (unsigned)(size_t)&sdbl[0][0], SCHUNK);
  for (int c = 0; c < nch; ++c) {
    if (issuer) {
      if (c + 1 < nch) {
        tdm_load_rows(dbl + (size_t)(t0 + (c + 1) * SCHUNK) * 48,
                      (unsigned)(size_t)&sdbl[(c + 1) & 1][0], SCHUNK);
        __builtin_amdgcn_s_wait_tensorcnt(1);  // chunk c done (in-order)
      } else {
        __builtin_amdgcn_s_wait_tensorcnt(0);
      }
    }
    __syncthreads();
    const float* __restrict__ rows = &sdbl[c & 1][0];
    for (int j = 0; j < SCHUNK; ++j) {
      int t = t0 + c * SCHUNK + j;
      float dlt = dy[t * 512 + d];
      float uv = (float)uact[t * 512 + d];
      __builtin_prefetch(&dy[(t + 8) * 512 + d], 0, 1);
      const float* Brow = rows + j * 48 + 16;
      const float* Crow = rows + j * 48 + 32;
      float du = dlt * uv;
      float y = 0.f;
#pragma unroll
      for (int n = 0; n < NST; n++) {
        float hn = __expf(dlt * A[n]) * h[n] + du * Brow[n];
        h[n] = hn;
        y += hn * Crow[n];
      }
      dy[t * 512 + d] = y;
    }
    __syncthreads();                           // buffer reuse fence
  }
}

// ---------------- 7: gate: yg = (y + u*D) * silu(z) -------------------------
__global__ void gate_kernel(const float* __restrict__ y, const __bf16* __restrict__ uact,
                            const __bf16* __restrict__ xz, const float* __restrict__ Dp,
                            __bf16* __restrict__ yg) {
  int idx = blockIdx.x * 256 + threadIdx.x;
  int d = idx & 511, t = idx >> 9;
  float z = (float)xz[t * 1024 + 512 + d];
  float val = (y[idx] + (float)uact[idx] * Dp[d]) * (z * (1.f / (1.f + __expf(-z))));
  yg[idx] = (__bf16)val;
}

// ---------------- 8: out_proj GEMM h[T,256] = yg[T,512] * W[256,512]^T ------
// 4 waves/block, each wave owns 4 N-tiles (A reuse).
__global__ void outproj_gemm(const __bf16* __restrict__ yg, const __bf16* __restrict__ W,
                             float* __restrict__ h) {
  int lane = threadIdx.x & 31, wave = threadIdx.x >> 5;
  int row0 = blockIdx.x * 16;
  int n0 = wave * 64;
  v8f acc[4] = {};
  gemm_multi<4>(yg, 512, W, 512, row0, n0, 512, lane, acc);
  int rb = (lane >> 4) * 8;
#pragma unroll
  for (int t = 0; t < 4; t++) {
    int col = n0 + 16 * t + (lane & 15);
#pragma unroll
    for (int r = 0; r < 8; r++)
      h[(row0 + rb + r) * 256 + col] = acc[t][r];
  }
}

// ---------------- 9: SE mean over tokens (atomic partial sums) --------------
__global__ void se_mean(const float* __restrict__ h, float* __restrict__ s) {
  int b = blockIdx.y, chunk = blockIdx.x, c = threadIdx.x;
  int t0 = b * 4096 + chunk * 256;
  float acc = 0.f;
  for (int i = 0; i < 256; i++) acc += h[(t0 + i) * 256 + c];
  atomicAdd(&s[b * 256 + c], acc * (1.f / 4096.f));
}

// ---------------- 10: SE FCs: g = sigmoid(W2 relu(W1 s)) --------------------
__global__ void se_fc(const float* __restrict__ s, const float* __restrict__ w1,
                      const float* __restrict__ w2, float* __restrict__ g) {
  int b = blockIdx.x, tid = threadIdx.x;
  __shared__ float sv[256];
  __shared__ float r[16];
  sv[tid] = s[b * 256 + tid];
  __syncthreads();
  if (tid < 16) {
    float acc = 0.f;
    for (int c = 0; c < 256; c++) acc += sv[c] * w1[tid * 256 + c];
    r[tid] = fmaxf(acc, 0.f);
  }
  __syncthreads();
  float acc = 0.f;
#pragma unroll
  for (int jj = 0; jj < 16; jj++) acc += r[jj] * w2[tid * 16 + jj];
  g[b * 256 + tid] = 1.f / (1.f + __expf(-acc));
}

// ---------------- 11: residual + SE scale + transpose back to [B,C,H,W] -----
__global__ void fuse_out(const float* __restrict__ x, const float* __restrict__ h,
                         const float* __restrict__ g, float* __restrict__ out) {
  __shared__ float tile[32][33];
  int b = blockIdx.z, c0 = blockIdx.y * 32, hw0 = blockIdx.x * 32;
  int tx = threadIdx.x, ty = threadIdx.y;
  int c = c0 + tx, hw = hw0 + ty;
  tile[ty][tx] = h[(b * 4096 + hw) * 256 + c] * g[b * 256 + c];
  __syncthreads();
  c = c0 + ty; hw = hw0 + tx;
  int gi = (b * 256 + c) * 4096 + hw;
  out[gi] = x[gi] + tile[tx][ty];
}

// ---------------- launch --------------------------------------------------
extern "C" void kernel_launch(void* const* d_in, const int* in_sizes, int n_in,
                              void* d_out, int out_size, void* d_ws, size_t ws_size,
                              hipStream_t stream) {
  const float* x      = (const float*)d_in[0];
  const float* ln_w   = (const float*)d_in[1];
  const float* mn_w   = (const float*)d_in[2];
  const float* mn_b   = (const float*)d_in[3];
  const float* inpw   = (const float*)d_in[4];
  const float* convw  = (const float*)d_in[5];
  const float* convb  = (const float*)d_in[6];
  const float* xprojw = (const float*)d_in[7];
  const float* dtw    = (const float*)d_in[8];
  const float* dtb    = (const float*)d_in[9];
  const float* A_log  = (const float*)d_in[10];
  const float* Dp     = (const float*)d_in[11];
  const float* outw   = (const float*)d_in[12];
  const float* sew1   = (const float*)d_in[13];
  const float* sew2   = (const float*)d_in[14];
  float* out = (float*)d_out;

  char* ws = (char*)d_ws;
  size_t off = 0;
  auto wsalloc = [&](size_t bytes) -> void* {
    void* p = ws + off;
    off = (off + bytes + 255) & ~(size_t)255;
    return p;
  };
  __bf16* xn    = (__bf16*)wsalloc((size_t)TTOK * 256 * 2);      //  8 MB
  __bf16* xz    = (__bf16*)wsalloc((size_t)TTOK * 1024 * 2);     // 32 MB
  __bf16* uact  = (__bf16*)wsalloc((size_t)TTOK * 512 * 2);      // 16 MB
  float*  dbl   = (float*) wsalloc((size_t)TTOK * 48 * 4);       //  3 MB
  float*  dy    = (float*) wsalloc((size_t)TTOK * 512 * 4);      // 32 MB (delta -> y)
  __bf16* yg    = (__bf16*)wsalloc((size_t)TTOK * 512 * 2);      // 16 MB
  float*  hbuf  = (float*) wsalloc((size_t)TTOK * 256 * 4);      // 16 MB
  float*  sbuf  = (float*) wsalloc(1024 * 4);
  float*  gbuf  = (float*) wsalloc(1024 * 4);
  __bf16* wb_in = (__bf16*)wsalloc(1024 * 256 * 2);
  __bf16* wb_x  = (__bf16*)wsalloc(48 * 512 * 2);
  __bf16* wb_dt = (__bf16*)wsalloc(512 * 16 * 2);
  __bf16* wb_o  = (__bf16*)wsalloc(256 * 512 * 2);

  // weights -> bf16
  cvt_f32_bf16<<<(1024 * 256 + 255) / 256, 256, 0, stream>>>(inpw, wb_in, 1024 * 256);
  cvt_f32_bf16<<<(48 * 512 + 255) / 256, 256, 0, stream>>>(xprojw, wb_x, 48 * 512);
  cvt_f32_bf16<<<(512 * 16 + 255) / 256, 256, 0, stream>>>(dtw, wb_dt, 512 * 16);
  cvt_f32_bf16<<<(256 * 512 + 255) / 256, 256, 0, stream>>>(outw, wb_o, 256 * 512);
  zero_f32<<<4, 256, 0, stream>>>(sbuf, 1024);

  // pipeline
  ln2_kernel<<<dim3(LSEQ / 32, BB), 256, 0, stream>>>(x, ln_w, mn_w, mn_b, xn);
  inproj_gemm<<<dim3(TTOK / 16, 2), 256, 0, stream>>>(xn, wb_in, xz);
  conv_silu<<<(TTOK * 512) / 256, 256, 0, stream>>>(xz, convw, convb, uact);
  xproj_gemm<<<TTOK / 16, 32, 0, stream>>>(uact, wb_x, dbl);
  dtproj_gemm<<<dim3(TTOK / 16, 8), 32, 0, stream>>>(dbl, wb_dt, dtb, dy);
  scan_kernel<<<BB * 2, 256, 0, stream>>>(A_log, uact, dbl, dy);
  gate_kernel<<<(TTOK * 512) / 256, 256, 0, stream>>>(dy, uact, xz, Dp, yg);
  outproj_gemm<<<TTOK / 16, 128, 0, stream>>>(yg, wb_o, hbuf);
  se_mean<<<dim3(16, BB), 256, 0, stream>>>(hbuf, sbuf);
  se_fc<<<BB, 256, 0, stream>>>(sbuf, sew1, sew2, gbuf);
  fuse_out<<<dim3(LSEQ / 32, CDIM / 32, BB), dim3(32, 32), 0, stream>>>(x, hbuf, gbuf, out);

  (void)in_sizes; (void)n_in; (void)out_size; (void)ws_size;
}
